// Attention_7679401525250
// MI455X (gfx1250) — matrix operand order repaired and checked by reading, not compiled, not run on hardware
//
#include <hip/hip_runtime.h>
#include <hip/hip_bf16.h>

// ---------------------------------------------------------------------------
// MI455X (gfx1250) attention pipeline.
//   wave32 + v_wmma_f32_16x16x32_f16 everywhere; the two dense GEMMs use the
//   Tensor Data Mover (tensor_load_to_lds + s_wait_tensorcnt) to double-buffer
//   A/B f16 tiles in LDS, shared by all 8 waves of the workgroup.
//   sched_group_barrier pins "all fragment loads, then all WMMAs" so the
//   waitcnt pass emits staged s_wait_dscnt instead of full drains per WMMA.
// B=2, S=2048, DIM=2048, NH=32, NKV=8, HD=64, N_REP=4.
// ---------------------------------------------------------------------------

typedef _Float16 half_t;
typedef __attribute__((ext_vector_type(16))) _Float16 v16h;
typedef __attribute__((ext_vector_type(8)))  _Float16 v8h;
typedef __attribute__((ext_vector_type(8)))  float    v8f;
typedef __attribute__((ext_vector_type(4)))  unsigned int u32x4;
typedef __attribute__((ext_vector_type(8)))  int      i32x8;
typedef __attribute__((ext_vector_type(4)))  int      i32x4;

#define BQ   2
#define SQ   2048
#define DIMQ 2048
#define NHQ  32
#define NKVQ 8
#define HDQ  64
#define MROWS (BQ * SQ)                 // 4096
#define NQKV  (DIMQ + 2 * NKVQ * HDQ)   // 3072

// Scheduling-group hints (no-ops if unsupported).
// mask: 0x008 = MFMA/WMMA, 0x020 = VMEM read, 0x100 = DS read.
#if __has_builtin(__builtin_amdgcn_sched_group_barrier)
#define SCHED_GROUP(mask, size, id) __builtin_amdgcn_sched_group_barrier((mask), (size), (id))
#else
#define SCHED_GROUP(mask, size, id)
#endif

__device__ __forceinline__ v8f wmma_f16(v16h a, v16h b, v8f c) {
    return __builtin_amdgcn_wmma_f32_16x16x32_f16(
        /*neg_a=*/false, a, /*neg_b=*/false, b,
        /*c_mod=*/(short)0, c, /*reuse_a=*/false, /*reuse_b=*/false);
}

// A-fragment (16x32 f16): lane l holds row M=l%16; the 16 halves
// are A[M][16*half .. +15] contiguous  (half = l/16), matching B.
__device__ __forceinline__ v16h load_a_frag(const half_t* rowptr, int half_sel) {
    v8h a0 = *(const v8h*)(rowptr + 16 * half_sel);
    v8h a1 = *(const v8h*)(rowptr + 16 * half_sel + 8);
    v16h r;
#pragma unroll
    for (int i = 0; i < 8; ++i) { r[i] = a0[i]; r[i + 8] = a1[i]; }
    return r;
}

// B-fragment (32x16 f16): lane l holds col N=l%16, K = 16*(l/16)+0..15,
// contiguous when B is stored transposed ([N][K] row-major).
__device__ __forceinline__ v16h load_b_frag(const half_t* bt_row_k) {
    return *(const v16h*)bt_row_k;   // 32B contiguous
}

// ---------------------------------------------------------------------------
// TDM: 2-D tile DMA global(f16) -> LDS.  D# per ISA 8.3/8.4:
//   group0: count=1, lds_addr, global_addr(57b), type=2 ("image")
//   group1: wg_mask=0, data_size=1 (2B), tensor_dim0/1, tile_dim0/1,
//           tensor_dim0_stride; groups 2/3 zero (<=2D tensor).
// ---------------------------------------------------------------------------
__device__ __forceinline__ void tdm_load_tile_2d(unsigned lds_byte_addr,
                                                 const half_t* gptr,
                                                 unsigned tileK, unsigned tileRows,
                                                 unsigned rowStrideElems,
                                                 unsigned tensorDim0, unsigned tensorDim1) {
    unsigned long long ga = (unsigned long long)(size_t)gptr;
    u32x4 g0;
    g0[0] = 1u;                                          // count=1, user mode
    g0[1] = lds_byte_addr;                               // LDS destination
    g0[2] = (unsigned)(ga & 0xFFFFFFFFu);                // global_addr[31:0]
    g0[3] = (unsigned)((ga >> 32) & 0x01FFFFFFu) | (2u << 30);  // [56:32] | type=2
    i32x8 g1;
    g1[0] = (int)(1u << 16);                             // data_size=1 (2 bytes)
    g1[1] = (int)((tensorDim0 & 0xFFFFu) << 16);         // tensor_dim0[15:0]
    g1[2] = (int)(((tensorDim0 >> 16) & 0xFFFFu) | ((tensorDim1 & 0xFFFFu) << 16));
    g1[3] = (int)(((tensorDim1 >> 16) & 0xFFFFu) | ((tileK & 0xFFFFu) << 16)); // tile_dim0
    g1[4] = (int)(tileRows & 0xFFFFu);                   // tile_dim1 (tile_dim2=0)
    g1[5] = (int)rowStrideElems;                         // tensor_dim0_stride[31:0]
    g1[6] = 0;                                           // stride msbs / dim1_stride
    g1[7] = 0;
    i32x4 z4 = {0, 0, 0, 0};
#if defined(__clang_major__) && __clang_major__ >= 23
    i32x8 z8 = {0, 0, 0, 0, 0, 0, 0, 0};
    __builtin_amdgcn_tensor_load_to_lds(g0, g1, z4, z4, z8, 0);
#else
    __builtin_amdgcn_tensor_load_to_lds(g0, g1, z4, z4, 0);   // ROCm 7.2 / clang-22
#endif
}

__device__ __forceinline__ unsigned lds_addr_of(const void* p) {
    return (unsigned)(size_t)p;       // flat LDS aperture: offset = addr[31:0]
}

// ---------------------------------------------------------------------------
// Kernel 1: x f32 -> f16
// ---------------------------------------------------------------------------
__global__ __launch_bounds__(256) void convert_x_kernel(const float* __restrict__ src,
                                                        half_t* __restrict__ dst) {
    int id = blockIdx.x * 256 + threadIdx.x;
    dst[id] = (half_t)src[id];
}

// ---------------------------------------------------------------------------
// Kernel 2: transpose + convert  src f32 [R][C]  ->  dst f16 [C][R]
// ---------------------------------------------------------------------------
__global__ __launch_bounds__(256) void transpose_cvt_kernel(const float* __restrict__ src,
                                                            half_t* __restrict__ dst,
                                                            int R, int C) {
    __shared__ float tile[32][33];
    int c0 = blockIdx.x * 32, r0 = blockIdx.y * 32;
    int tx = threadIdx.x, ty = threadIdx.y;
#pragma unroll
    for (int i = ty; i < 32; i += 8)
        tile[i][tx] = src[(size_t)(r0 + i) * C + c0 + tx];
    __syncthreads();
#pragma unroll
    for (int i = ty; i < 32; i += 8)
        dst[(size_t)(c0 + i) * R + r0 + tx] = (half_t)tile[tx][i];
}

// ---------------------------------------------------------------------------
// Kernel 3: fused QKV projection GEMM + RoPE.  Cooperative 128x64 C tile:
//   TDM double-buffers A(128x32) and B(64x32) f16 tiles into LDS; each of the
//   8 waves owns a 16-row strip and shares the B tile.
// ---------------------------------------------------------------------------
__global__ __launch_bounds__(256) void qkv_rope_gemm_kernel(const half_t* __restrict__ x16,
                                                            const half_t* __restrict__ wT,
                                                            half_t* __restrict__ Qb,
                                                            half_t* __restrict__ Kb,
                                                            half_t* __restrict__ VTb) {
    __shared__ half_t aT[2][128][32];   // 16 KB
    __shared__ half_t bT[2][64][32];    //  8 KB

    const int wv   = threadIdx.x >> 5;
    const int lane = threadIdx.x & 31;
    const int hs   = lane >> 4;
    const int ln   = lane & 15;
    const int bn   = blockIdx.x % (NQKV / 64);   // 48 column tiles of 64
    const int bm   = blockIdx.x / (NQKV / 64);   // 32 row tiles of 128

    const half_t* aSrc = x16 + (size_t)(bm * 128) * DIMQ;
    const half_t* bSrc = wT  + (size_t)(bn * 64) * DIMQ;

    if (wv == 0) {   // prologue DMA for kc = 0 into buffer 0
        tdm_load_tile_2d(lds_addr_of(&aT[0][0][0]), aSrc, 32, 128, DIMQ, DIMQ, MROWS);
        tdm_load_tile_2d(lds_addr_of(&bT[0][0][0]), bSrc, 32, 64, DIMQ, DIMQ, NQKV);
    }

    const v8f vzero = {0, 0, 0, 0, 0, 0, 0, 0};
    v8f acc[4] = {vzero, vzero, vzero, vzero};
    int cur = 0;

    for (int kc = 0; kc < DIMQ; kc += 32) {
        if (wv == 0) __builtin_amdgcn_s_wait_tensorcnt(0);
        __syncthreads();                              // buffers[cur] ready, prev reads done
        if (wv == 0 && kc + 32 < DIMQ) {              // DMA next chunk into other buffer
            tdm_load_tile_2d(lds_addr_of(&aT[cur ^ 1][0][0]), aSrc + kc + 32,
                             32, 128, DIMQ, DIMQ, MROWS);
            tdm_load_tile_2d(lds_addr_of(&bT[cur ^ 1][0][0]), bSrc + kc + 32,
                             32, 64, DIMQ, DIMQ, NQKV);
        }
        // Batch all fragment loads, then run the WMMA group back-to-back.
        v16h a  = load_a_frag(&aT[cur][wv * 16 + ln][0], hs);
        v16h b0 = load_b_frag(&bT[cur][ 0 + ln][16 * hs]);
        v16h b1 = load_b_frag(&bT[cur][16 + ln][16 * hs]);
        v16h b2 = load_b_frag(&bT[cur][32 + ln][16 * hs]);
        v16h b3 = load_b_frag(&bT[cur][48 + ln][16 * hs]);
        acc[0] = wmma_f16(a, b0, acc[0]);
        acc[1] = wmma_f16(a, b1, acc[1]);
        acc[2] = wmma_f16(a, b2, acc[2]);
        acc[3] = wmma_f16(a, b3, acc[3]);
        // Pin the schedule: 10 DS reads first (in-order returns -> staged
        // s_wait_dscnt), then the 4 WMMAs back-to-back.
        SCHED_GROUP(0x100, 10, 0);
        SCHED_GROUP(0x008, 4, 0);
        cur ^= 1;
    }

    // Epilogue: RoPE via adjacent-lane exchange (even/odd head dims are
    // adjacent N columns == adjacent lanes in the C layout).
#pragma unroll
    for (int j = 0; j < 4; ++j) {
        const int n = bn * 64 + j * 16 + ln;          // global output column
        const int d = n & 63;                         // head-dim index (q & k)
        const float inv_freq = __expf(-(float)(d & ~1) * (9.210340371976184f / (float)HDQ));
#pragma unroll
        for (int r = 0; r < 8; ++r) {
            float v = acc[j][r];
            float other = __shfl_xor(v, 1, 32);       // partner column (d^1)
            const int m = bm * 128 + wv * 16 + r + 8 * hs;
            const int b = m >> 11;                    // m / S
            const int s = m & (SQ - 1);
            if (n < DIMQ + NKVQ * HDQ) {              // Q or K: apply RoPE
                float c, sn;
                __sincosf((float)s * inv_freq, &c, &sn);
                float res = (d & 1) ? (other * sn + v * c)   // odd  = x0*s + x1*c
                                    : (v * c - other * sn);  // even = x0*c - x1*s
                if (n < DIMQ) {
                    const int h = n >> 6;
                    Qb[((size_t)(b * NHQ + h) * SQ + s) * HDQ + d] = (half_t)res;
                } else {
                    const int h = (n - DIMQ) >> 6;
                    Kb[((size_t)(b * NKVQ + h) * SQ + s) * HDQ + d] = (half_t)res;
                }
            } else {                                  // V: store transposed
                const int nv = n - (DIMQ + NKVQ * HDQ);
                const int h = nv >> 6, dv = nv & 63;
                VTb[((size_t)(b * NKVQ + h) * HDQ + dv) * SQ + s] = (half_t)v;
            }
        }
    }
}

// ---------------------------------------------------------------------------
// Kernel 4: flash attention (online softmax), GQA 4:1, causal.
//   Per-wave: 16 q rows; 32 k cols/iter -> 4 score WMMAs, shfl-butterfly
//   softmax, P staged through per-wave LDS (C->A layout), 4 PV WMMAs.
//   (Per-wave causal trip counts forbid workgroup barriers here, so this
//   kernel keeps direct global loads + prefetch instead of TDM.)
// ---------------------------------------------------------------------------
__global__ __launch_bounds__(256) void flash_attn_kernel(const half_t* __restrict__ Qb,
                                                         const half_t* __restrict__ Kb,
                                                         const half_t* __restrict__ VTb,
                                                         half_t* __restrict__ Ob) {
    __shared__ half_t plds[8][16][32];                // 8 KB, one tile per wave

    const int wv   = threadIdx.x >> 5;
    const int lane = threadIdx.x & 31;
    const int hs   = lane >> 4;
    const int ln   = lane & 15;

    const int blk  = blockIdx.x;                      // ((b*NH + h)*16 + qblk)
    const int qblk = blk & 15;
    const int h    = (blk >> 4) & (NHQ - 1);
    const int b    = blk >> 9;
    const int qbase = qblk * 128 + wv * 16;
    const int hk   = h >> 2;                          // N_REP = 4

    const half_t* qptr = Qb  + (size_t)(b * NHQ  + h)  * SQ * HDQ;
    const half_t* kptr = Kb  + (size_t)(b * NKVQ + hk) * SQ * HDQ;
    const half_t* vtp  = VTb + (size_t)(b * NKVQ + hk) * HDQ * SQ;

    // Q fragments resident in registers for the whole KV sweep.
    const half_t* qrow = qptr + (size_t)(qbase + ln) * HDQ;
    v16h qa[2];
    qa[0] = load_a_frag(qrow, hs);
    qa[1] = load_a_frag(qrow + 32, hs);

    const v8f vzero = {0, 0, 0, 0, 0, 0, 0, 0};
    v8f o[4] = {vzero, vzero, vzero, vzero};
    float m_[8], l_[8];
#pragma unroll
    for (int r = 0; r < 8; ++r) { m_[r] = -1e30f; l_[r] = 0.0f; }

    const float scale = 0.125f;                        // 1/sqrt(64)

    for (int kb = 0; kb < qbase + 16; kb += 32) {
        if (kb + 32 < qbase + 16)                      // hint next K tile
            __builtin_prefetch(kptr + (size_t)(kb + 32 + ln) * HDQ, 0, 0);

        // ---- scores: S = Q K^T for 32 key columns (two 16-col tiles) ----
        v16h k00 = load_b_frag(kptr + (size_t)(kb +      ln) * HDQ +  0 + 16 * hs);
        v16h k01 = load_b_frag(kptr + (size_t)(kb + 16 + ln) * HDQ +  0 + 16 * hs);
        v16h k10 = load_b_frag(kptr + (size_t)(kb +      ln) * HDQ + 32 + 16 * hs);
        v16h k11 = load_b_frag(kptr + (size_t)(kb + 16 + ln) * HDQ + 32 + 16 * hs);
        v8f s0 = vzero, s1 = vzero;
        s0 = wmma_f16(qa[0], k00, s0);
        s1 = wmma_f16(qa[0], k01, s1);
        s0 = wmma_f16(qa[1], k10, s0);
        s1 = wmma_f16(qa[1], k11, s1);
        SCHED_GROUP(0x020, 8, 1);    // 8 VMEM reads (4 v16h K frags)
        SCHED_GROUP(0x008, 4, 1);    // 4 score WMMAs

        const bool need_mask = (kb + 31 > qbase);
        float alpha[8];
#pragma unroll
        for (int r = 0; r < 8; ++r) {
            float a0 = s0[r] * scale, a1 = s1[r] * scale;
            if (need_mask) {
                const int row = qbase + r + 8 * hs;
                if (kb + ln      > row) a0 = -1e30f;
                if (kb + 16 + ln > row) a1 = -1e30f;
            }
            // row max across the 32 columns held by this 16-lane half
            float mx = fmaxf(a0, a1);
#pragma unroll
            for (int off = 1; off < 16; off <<= 1)
                mx = fmaxf(mx, __shfl_xor(mx, off, 32));
            const float mnew = fmaxf(m_[r], mx);
            alpha[r] = __expf(m_[r] - mnew);
            a0 = __expf(a0 - mnew);
            a1 = __expf(a1 - mnew);
            s0[r] = a0; s1[r] = a1;
            float rs = a0 + a1;
#pragma unroll
            for (int off = 1; off < 16; off <<= 1)
                rs += __shfl_xor(rs, off, 32);
            l_[r] = l_[r] * alpha[r] + rs;
            m_[r] = mnew;
        }
#pragma unroll
        for (int j = 0; j < 4; ++j)
#pragma unroll
            for (int r = 0; r < 8; ++r) o[j][r] *= alpha[r];

        // ---- P: C-layout -> A-layout via per-wave LDS tile ----
#pragma unroll
        for (int r = 0; r < 8; ++r) {
            plds[wv][r + 8 * hs][ln]      = (half_t)s0[r];
            plds[wv][r + 8 * hs][16 + ln] = (half_t)s1[r];
        }
        __builtin_amdgcn_wave_barrier();   // DS ops are in-order per wave
        v16h pa = load_a_frag(&plds[wv][ln][0], hs);

        // ---- O += P * V  (V transposed -> contiguous B fragments) ----
        v16h vb0 = load_b_frag(vtp + (size_t)( 0 + ln) * SQ + kb + 16 * hs);
        v16h vb1 = load_b_frag(vtp + (size_t)(16 + ln) * SQ + kb + 16 * hs);
        v16h vb2 = load_b_frag(vtp + (size_t)(32 + ln) * SQ + kb + 16 * hs);
        v16h vb3 = load_b_frag(vtp + (size_t)(48 + ln) * SQ + kb + 16 * hs);
        o[0] = wmma_f16(pa, vb0, o[0]);
        o[1] = wmma_f16(pa, vb1, o[1]);
        o[2] = wmma_f16(pa, vb2, o[2]);
        o[3] = wmma_f16(pa, vb3, o[3]);
        SCHED_GROUP(0x100, 2, 2);    // 2 DS reads (P A-frag)
        SCHED_GROUP(0x020, 8, 2);    // 8 VMEM reads (4 v16h V frags)
        SCHED_GROUP(0x008, 4, 2);    // 4 PV WMMAs
        __builtin_amdgcn_wave_barrier();
    }

    // ---- finalize: O /= rowsum, store as f16 into [B, S, NH*HD] ----
#pragma unroll
    for (int r = 0; r < 8; ++r) {
        const float inv = 1.0f / l_[r];
        const int s = qbase + r + 8 * hs;
        const size_t base = ((size_t)b * SQ + s) * DIMQ + h * HDQ;
#pragma unroll
        for (int j = 0; j < 4; ++j)
            Ob[base + j * 16 + ln] = (half_t)(o[j][r] * inv);
    }
}

// ---------------------------------------------------------------------------
// Kernel 5: output projection  out = O(f16) @ wo -> f32, cooperative 128x64
//   tile with TDM-fed LDS double buffers (same scheme as kernel 3).
// ---------------------------------------------------------------------------
__global__ __launch_bounds__(256) void out_gemm_kernel(const half_t* __restrict__ Ob,
                                                       const half_t* __restrict__ woT,
                                                       float* __restrict__ out) {
    __shared__ half_t aT[2][128][32];
    __shared__ half_t bT[2][64][32];

    const int wv   = threadIdx.x >> 5;
    const int lane = threadIdx.x & 31;
    const int hs   = lane >> 4;
    const int ln   = lane & 15;
    const int bn   = blockIdx.x % (DIMQ / 64);    // 32 column tiles
    const int bm   = blockIdx.x / (DIMQ / 64);    // 32 row tiles of 128

    const half_t* aSrc = Ob  + (size_t)(bm * 128) * DIMQ;
    const half_t* bSrc = woT + (size_t)(bn * 64) * DIMQ;

    if (wv == 0) {
        tdm_load_tile_2d(lds_addr_of(&aT[0][0][0]), aSrc, 32, 128, DIMQ, DIMQ, MROWS);
        tdm_load_tile_2d(lds_addr_of(&bT[0][0][0]), bSrc, 32, 64, DIMQ, DIMQ, DIMQ);
    }

    const v8f vzero = {0, 0, 0, 0, 0, 0, 0, 0};
    v8f acc[4] = {vzero, vzero, vzero, vzero};
    int cur = 0;

    for (int kc = 0; kc < DIMQ; kc += 32) {
        if (wv == 0) __builtin_amdgcn_s_wait_tensorcnt(0);
        __syncthreads();
        if (wv == 0 && kc + 32 < DIMQ) {
            tdm_load_tile_2d(lds_addr_of(&aT[cur ^ 1][0][0]), aSrc + kc + 32,
                             32, 128, DIMQ, DIMQ, MROWS);
            tdm_load_tile_2d(lds_addr_of(&bT[cur ^ 1][0][0]), bSrc + kc + 32,
                             32, 64, DIMQ, DIMQ, DIMQ);
        }
        v16h a  = load_a_frag(&aT[cur][wv * 16 + ln][0], hs);
        v16h b0 = load_b_frag(&bT[cur][ 0 + ln][16 * hs]);
        v16h b1 = load_b_frag(&bT[cur][16 + ln][16 * hs]);
        v16h b2 = load_b_frag(&bT[cur][32 + ln][16 * hs]);
        v16h b3 = load_b_frag(&bT[cur][48 + ln][16 * hs]);
        acc[0] = wmma_f16(a, b0, acc[0]);
        acc[1] = wmma_f16(a, b1, acc[1]);
        acc[2] = wmma_f16(a, b2, acc[2]);
        acc[3] = wmma_f16(a, b3, acc[3]);
        SCHED_GROUP(0x100, 10, 0);
        SCHED_GROUP(0x008, 4, 0);
        cur ^= 1;
    }
#pragma unroll
    for (int r = 0; r < 8; ++r) {
        const size_t m = (size_t)(bm * 128 + wv * 16 + r + 8 * hs);
#pragma unroll
        for (int j = 0; j < 4; ++j)
            out[m * DIMQ + bn * 64 + j * 16 + ln] = acc[j][r];
    }
}

// ---------------------------------------------------------------------------
// Host-side launch sequencing (graph-capture safe: only kernel launches).
// ---------------------------------------------------------------------------
extern "C" void kernel_launch(void* const* d_in, const int* in_sizes, int n_in,
                              void* d_out, int out_size, void* d_ws, size_t ws_size,
                              hipStream_t stream) {
    (void)in_sizes; (void)n_in; (void)out_size; (void)ws_size;
    const float* x  = (const float*)d_in[0];
    const float* wq = (const float*)d_in[1];
    const float* wk = (const float*)d_in[2];
    const float* wv = (const float*)d_in[3];
    const float* wo = (const float*)d_in[4];
    float* out = (float*)d_out;

    // Workspace layout (f16 element offsets), ~80 MB total.
    half_t* ws  = (half_t*)d_ws;
    half_t* x16 = ws;                                            // [4096][2048]
    half_t* wT  = x16 + (size_t)MROWS * DIMQ;                    // [3072][2048]
    half_t* woT = wT  + (size_t)NQKV  * DIMQ;                    // [2048][2048]
    half_t* Qb  = woT + (size_t)DIMQ  * DIMQ;                    // [B][NH][S][HD]
    half_t* Kb  = Qb  + (size_t)BQ * NHQ  * SQ * HDQ;            // [B][NKV][S][HD]
    half_t* VTb = Kb  + (size_t)BQ * NKVQ * SQ * HDQ;            // [B][NKV][HD][S]
    half_t* Ob  = VTb + (size_t)BQ * NKVQ * HDQ * SQ;            // [4096][2048]

    // 1) activations -> f16
    convert_x_kernel<<<(MROWS * DIMQ) / 256, 256, 0, stream>>>(x, x16);

    // 2) weights -> f16, transposed to [N][K] for contiguous B fragments
    dim3 tb(32, 8);
    transpose_cvt_kernel<<<dim3(DIMQ / 32, DIMQ / 32), tb, 0, stream>>>(wq, wT, DIMQ, DIMQ);
    transpose_cvt_kernel<<<dim3(512 / 32, DIMQ / 32), tb, 0, stream>>>(wk, wT + (size_t)DIMQ * DIMQ, DIMQ, 512);
    transpose_cvt_kernel<<<dim3(512 / 32, DIMQ / 32), tb, 0, stream>>>(wv, wT + (size_t)(DIMQ + 512) * DIMQ, DIMQ, 512);
    transpose_cvt_kernel<<<dim3(DIMQ / 32, DIMQ / 32), tb, 0, stream>>>(wo, woT, DIMQ, DIMQ);

    // 3) fused QKV projection + RoPE: 32 x 48 cooperative 128x64 tiles
    qkv_rope_gemm_kernel<<<(MROWS / 128) * (NQKV / 64), 256, 0, stream>>>(x16, wT, Qb, Kb, VTb);

    // 4) flash attention: B*NH*(S/128) blocks, 8 waves each
    flash_attn_kernel<<<BQ * NHQ * (SQ / 128), 256, 0, stream>>>(Qb, Kb, VTb, Ob);

    // 5) output projection: 32 x 32 cooperative 128x64 tiles
    out_gemm_kernel<<<(MROWS / 128) * (DIMQ / 64), 256, 0, stream>>>(Ob, woT, out);
}
